// SkyCubeMap_codebook_54322746360436
// MI455X (gfx1250) — matrix operand order repaired and checked by reading, not compiled, not run on hardware
//
#include <hip/hip_runtime.h>
#include <hip/hip_bf16.h>
#include <math.h>

typedef __attribute__((ext_vector_type(16))) _Float16 v16h;
typedef __attribute__((ext_vector_type(8)))  float    v8f;

#define RES   1024
#define FEAT  12
#define CB    32
#define H_IMG 1080
#define W_IMG 1920
#define NPIX  (H_IMG * W_IMG)
#define HSTRIDE 36   // halfs per pixel row in LDS (72B: 8B-aligned, bank-friendly)

__device__ __forceinline__ float relu_f(float x) { return x > 0.f ? x : 0.f; }
__device__ __forceinline__ float sigmoid_f(float x) { return 1.f / (1.f + __expf(-x)); }

// Same-wave LDS ordering only: compiler fence + s_wait_dscnt, no s_barrier.
// Safe because each wave reads only LDS rows written by its own lanes, and
// CDNA5 executes same-wave DS instructions in order.
__device__ __forceinline__ void wave_lds_fence() {
  __builtin_amdgcn_fence(__ATOMIC_ACQ_REL, "wavefront");
}

// D = A*B + C, f16 inputs, f32 accumulate (emits v_wmma_f32_16x16x32_f16)
__device__ __forceinline__ v8f wmma_f16(v16h a, v16h b, v8f c) {
  return __builtin_amdgcn_wmma_f32_16x16x32_f16(
      /*neg_a=*/false, a, /*neg_b=*/false, b,
      /*c_mod=*/(short)0, c, /*reuse_a=*/false, /*reuse_b=*/false);
}

// Build one 32x16 B-matrix fragment (K=32, N=16 tile at column offset noff)
// from a row-major f32 weight matrix W[nrows][ncols].  Contiguous K layout (matches load_afrag):
// lanes 0-15 hold K=0..15 for N=lane; lanes 16-31 hold K=16..31 for N=lane-16.
__device__ __forceinline__ v16h load_bfrag(const float* __restrict__ W,
                                           int ncols, int nrows, int noff, int lane) {
  v16h b;
  int n = (lane & 15) + noff;
  int kbase = (lane < 16) ? 0 : 16;
#pragma unroll
  for (int e = 0; e < 16; ++e) {
    int k = kbase + e;
    float w = (k < nrows && n < ncols) ? W[k * ncols + n] : 0.f;
    b[e] = (_Float16)w;
  }
  return b;
}

// Load a 16x32 A-matrix fragment (f16) from LDS rows [base .. base+15].
// Layout: lane m=lane&15 is row M; lanes<16 hold K=0..15, lanes>=16 hold K=16..31
// (contiguous, matching load_bfrag).
__device__ __forceinline__ v16h load_afrag(const _Float16* __restrict__ sh,
                                           int rowbase, int lane) {
  const _Float16* row = sh + (rowbase + (lane & 15)) * HSTRIDE;
  int khi = (lane < 16) ? 0 : 16;
  v16h a;
#pragma unroll
  for (int e = 0; e < 16; ++e) {
    int v = e >> 1, p = e & 1;
    int kb = (v < 4) ? (2 * v) : (16 + 2 * (v - 4));
    a[e] = row[e + khi]; (void)kb; (void)p;
  }
  return a;
}

__global__ __launch_bounds__(256) void sky_fused_kernel(
    const float* __restrict__ rays_d, const unsigned char* __restrict__ mask,
    const float* __restrict__ cube, const float* __restrict__ cbook,
    const float* __restrict__ W1, const float* __restrict__ b1,
    const float* __restrict__ W2, const float* __restrict__ b2,
    const float* __restrict__ W3, const float* __restrict__ b3,
    float* __restrict__ out) {
  __shared__ float    s_cb[CB * FEAT];
  __shared__ float    s_cbn[CB];
  __shared__ _Float16 s_h[256 * HSTRIDE];

  const int tid  = threadIdx.x;
  const int lane = tid & 31;
  const int wave = tid >> 5;
  const int pix  = blockIdx.x * 256 + tid;

  // ---- stage codebook + its squared norms in LDS (cross-wave: barrier) ----
  for (int i = tid; i < CB * FEAT; i += 256) s_cb[i] = cbook[i];
  __syncthreads();
  if (tid < CB) {
    float s = 0.f;
#pragma unroll
    for (int k = 0; k < FEAT; ++k) { float v = s_cb[tid * FEAT + k]; s += v * v; }
    s_cbn[tid] = s;
  }

  // ---- per-wave weight fragments (tiny, L2-resident) ----------------------
  v16h bW1_0 = load_bfrag(W1, 32, 15, 0, lane);
  v16h bW1_1 = load_bfrag(W1, 32, 15, 16, lane);
  v16h bW2_0 = load_bfrag(W2, 32, 32, 0, lane);
  v16h bW2_1 = load_bfrag(W2, 32, 32, 16, lane);
  v16h bW3_0 = load_bfrag(W3, 3, 32, 0, lane);
  const int ncol = lane & 15;
  const float bias1_0 = b1[ncol], bias1_1 = b1[ncol + 16];
  const float bias2_0 = b2[ncol], bias2_1 = b2[ncol + 16];
  const float bias3   = (ncol < 3) ? b3[ncol] : 0.f;

  // ---- cube-map face select + bilinear sample -----------------------------
  float dx = rays_d[pix * 3 + 0];
  float dy = rays_d[pix * 3 + 1];
  float dz = rays_d[pix * 3 + 2];
  float ax = fabsf(dx), ay = fabsf(dy), az = fabsf(dz);
  bool is_x = (ax >= ay) && (ax >= az);
  bool is_y = !is_x && (ay >= az);
  int face; float ma, u, v;
  if (is_x)      { face = (dx >= 0.f) ? 0 : 1; ma = ax; u = (dx >= 0.f) ? -dz : dz; v = -dy; }
  else if (is_y) { face = (dy >= 0.f) ? 2 : 3; ma = ay; u = dx; v = (dy >= 0.f) ? dz : -dz; }
  else           { face = (dz >= 0.f) ? 4 : 5; ma = az; u = (dz >= 0.f) ? dx : -dx; v = -dy; }
  const float eps = 1e-9f;
  float rinv = 1.f / (ma + eps);
  float s = (u * rinv + 1.f) * 0.5f * (float)RES - 0.5f;
  float t = (v * rinv + 1.f) * 0.5f * (float)RES - 0.5f;
  float x0f = floorf(s), y0f = floorf(t);
  float fx = s - x0f, fy = t - y0f;
  int x0 = min(max((int)x0f, 0), RES - 1);
  int x1 = min(x0 + 1, RES - 1);
  int y0 = min(max((int)y0f, 0), RES - 1);
  int y1 = min(y0 + 1, RES - 1);

  const float* fb = cube + (size_t)face * RES * RES * FEAT;
  const float4* p00 = reinterpret_cast<const float4*>(fb + ((size_t)y0 * RES + x0) * FEAT);
  const float4* p01 = reinterpret_cast<const float4*>(fb + ((size_t)y0 * RES + x1) * FEAT);
  const float4* p10 = reinterpret_cast<const float4*>(fb + ((size_t)y1 * RES + x0) * FEAT);
  const float4* p11 = reinterpret_cast<const float4*>(fb + ((size_t)y1 * RES + x1) * FEAT);
  float w00 = (1.f - fx) * (1.f - fy), w01 = fx * (1.f - fy);
  float w10 = (1.f - fx) * fy,         w11 = fx * fy;

  float feat[FEAT];
#pragma unroll
  for (int i = 0; i < 3; ++i) {             // 12 channels = 3x float4 per corner
    float4 a0 = p00[i], a1 = p01[i], a2 = p10[i], a3 = p11[i];
    feat[4 * i + 0] = a0.x * w00 + a1.x * w01 + a2.x * w10 + a3.x * w11;
    feat[4 * i + 1] = a0.y * w00 + a1.y * w01 + a2.y * w10 + a3.y * w11;
    feat[4 * i + 2] = a0.z * w00 + a1.z * w01 + a2.z * w10 + a3.z * w11;
    feat[4 * i + 3] = a0.w * w00 + a1.w * w01 + a2.w * w10 + a3.w * w11;
  }

  __syncthreads();   // codebook norms ready (cross-wave)

  // ---- codebook argmin (||f||^2 constant -> minimize ||c||^2 - 2 f.c) ----
  int best = 0; float bestd = 3.0e38f;
#pragma unroll 4
  for (int j = 0; j < CB; ++j) {
    float dot = 0.f;
#pragma unroll
    for (int k = 0; k < FEAT; ++k) dot += feat[k] * s_cb[j * FEAT + k];
    float dj = s_cbn[j] - 2.f * dot;
    if (dj < bestd) { bestd = dj; best = j; }
  }

  // forward value of the straight-through quantizer is exactly cb[best]
  _Float16* hp = &s_h[tid * HSTRIDE];
#pragma unroll
  for (int k = 0; k < FEAT; ++k) hp[k] = (_Float16)s_cb[best * FEAT + k];
  hp[12] = (_Float16)dx; hp[13] = (_Float16)dy; hp[14] = (_Float16)dz;
#pragma unroll
  for (int k = 15; k < 32; ++k) hp[k] = (_Float16)0.f;

  wave_lds_fence();   // own-wave rows only: no workgroup barrier needed

  // ---- MLP on the matrix pipe: each wave = 2 M-tiles of 16 pixels ---------
  const int wb   = wave * 32;
  const int mrow = (lane < 16) ? 0 : 8;

#pragma unroll
  for (int mt = 0; mt < 2; ++mt) {
    const int rb = wb + mt * 16;

    // layer 1: [16x32(f16)] x [32x32] -> relu -> LDS (overwrites own rows)
    v16h a1 = load_afrag(s_h, rb, lane);
    v8f c0 = {}, c1 = {};
    c0 = wmma_f16(a1, bW1_0, c0);
    c1 = wmma_f16(a1, bW1_1, c1);
    _Float16* baseh = s_h + rb * HSTRIDE;
#pragma unroll
    for (int r = 0; r < 8; ++r) {
      baseh[(mrow + r) * HSTRIDE + ncol]      = (_Float16)relu_f(c0[r] + bias1_0);
      baseh[(mrow + r) * HSTRIDE + ncol + 16] = (_Float16)relu_f(c1[r] + bias1_1);
    }
    wave_lds_fence();

    // layer 2: 32 -> 32, relu
    v16h a2 = load_afrag(s_h, rb, lane);
    v8f d0 = {}, d1 = {};
    d0 = wmma_f16(a2, bW2_0, d0);
    d1 = wmma_f16(a2, bW2_1, d1);
#pragma unroll
    for (int r = 0; r < 8; ++r) {
      baseh[(mrow + r) * HSTRIDE + ncol]      = (_Float16)relu_f(d0[r] + bias2_0);
      baseh[(mrow + r) * HSTRIDE + ncol + 16] = (_Float16)relu_f(d1[r] + bias2_1);
    }
    wave_lds_fence();

    // layer 3: 32 -> 3 (padded to 16), sigmoid, masked channel-major store
    v16h a3 = load_afrag(s_h, rb, lane);
    v8f e0 = {};
    e0 = wmma_f16(a3, bW3_0, e0);
    if (ncol < 3) {                 // lanes 0,1,2 and 16,17,18 carry RGB cols
#pragma unroll
      for (int r = 0; r < 8; ++r) {
        int gp = blockIdx.x * 256 + rb + mrow + r;
        float val = sigmoid_f(e0[r] + bias3);
        val = mask[gp] ? val : 0.f;           // sigmoid already in [0,1]
        out[(size_t)ncol * NPIX + gp] = val;
      }
    }
    wave_lds_fence();
  }
}

extern "C" void kernel_launch(void* const* d_in, const int* in_sizes, int n_in,
                              void* d_out, int out_size, void* d_ws, size_t ws_size,
                              hipStream_t stream) {
  const float*         rays_d = (const float*)d_in[0];
  const unsigned char* mask   = (const unsigned char*)d_in[1];   // jax bool = 1 byte
  const float*         cube   = (const float*)d_in[2];
  const float*         cbook  = (const float*)d_in[3];
  const float*         W1     = (const float*)d_in[4];
  const float*         b1     = (const float*)d_in[5];
  const float*         W2     = (const float*)d_in[6];
  const float*         b2     = (const float*)d_in[7];
  const float*         W3     = (const float*)d_in[8];
  const float*         b3     = (const float*)d_in[9];
  float*               out    = (float*)d_out;

  const int nblocks = NPIX / 256;   // 2,073,600 / 256 = 8100 exactly
  sky_fused_kernel<<<nblocks, 256, 0, stream>>>(
      rays_d, mask, cube, cbook, W1, b1, W2, b2, W3, b3, out);
}
